// DotProductAttention_27968827031628
// MI455X (gfx1250) — matrix operand order repaired and checked by reading, not compiled, not run on hardware
//
#include <hip/hip_runtime.h>
#include <math.h>

// ---------------------------------------------------------------------------
// CDNA5 (gfx1250, wave32) bf16 WMMA attention:
//   convert f32->bf16 once -> proj GEMMs -> scores GEMM (f32 out) ->
//   row softmax -> attn@V GEMM. All GEMM inner loops are pure
//   global_load_b128 + v_wmma_f32_16x16x32_bf16.
// ---------------------------------------------------------------------------

typedef __bf16 bf16_t;
typedef bf16_t v16bf __attribute__((ext_vector_type(16)));
typedef float  v8f   __attribute__((ext_vector_type(8)));

union FragU {
  v16bf v;
  uint4 q[2];
};

// A fragment chunk layout: [p .. p+7] and [p+16 .. p+23] (16B each).
static __device__ __forceinline__ v16bf ldA(const bf16_t* p) {
  FragU u;
  u.q[0] = *(const uint4*)(p);
  u.q[1] = *(const uint4*)(p + 16);
  return u.v;
}

// B fragment chunk layout: [p .. p+7] and [p+16 .. p+23] (16B each).
static __device__ __forceinline__ v16bf ldB(const bf16_t* p) {
  FragU u;
  u.q[0] = *(const uint4*)(p);
  u.q[1] = *(const uint4*)(p + 16);
  return u.v;
}

// ---------------------------------------------------------------------------
// Elementwise f32 -> bf16 (8 elements / thread, b128 in / b128 out).
// ---------------------------------------------------------------------------
__global__ __launch_bounds__(256) void cvt_f32_bf16(const float* __restrict__ in,
                                                    bf16_t* __restrict__ out) {
  const long i = ((long)blockIdx.x * 256 + threadIdx.x) * 8;
  const float4 a = *(const float4*)(in + i);
  const float4 b = *(const float4*)(in + i + 4);
  union { uint4 q; bf16_t h[8]; } u;
  u.h[0] = (bf16_t)a.x; u.h[1] = (bf16_t)a.y;
  u.h[2] = (bf16_t)a.z; u.h[3] = (bf16_t)a.w;
  u.h[4] = (bf16_t)b.x; u.h[5] = (bf16_t)b.y;
  u.h[6] = (bf16_t)b.z; u.h[7] = (bf16_t)b.w;
  *(uint4*)(out + i) = u.q;
}

// ---------------------------------------------------------------------------
// C[M,N] = A[M,K] * Bt[N,K]^T (+ bias), batched over blockIdx.z. bf16 in.
// Block: 256 thr = 8 waves (wave32), 128x128 tile; wave = 32x64 = 2x4 tiles.
// Fragment layouts per ISA 7.12.2:
//   A 16x32: lane&15 = row M; lane half picks K {0..7,16..23} / {8..15,24..31}
//   B 32x16 from [N,K] operand: lane&15 = col N; lane half picks K {0..7,16..23} / {8..15,24..31}
//   C/D: VGPR r -> M = r + 8*(lane>=16), N = lane&15
// OUT_MODE: 0 = f32 row-major, 1 = bf16 row-major,
//           2 = bf16 transposed per Srows-row batch (C[b][col][s]).
// ---------------------------------------------------------------------------
template <bool BIAS, int OUT_MODE>
__global__ __launch_bounds__(256) void wmma_gemm_bt(
    const bf16_t* __restrict__ A, const bf16_t* __restrict__ Bt,
    const float* __restrict__ bias, void* __restrict__ Cp,
    int N, int K, long lda, long ldb, long ldc,
    long strideA, long strideB, long strideC, int Srows) {
  const int lane = threadIdx.x & 31;
  const int wid  = threadIdx.x >> 5;
  const int tileM = blockIdx.y * 128 + (wid >> 1) * 32;
  const int tileN = blockIdx.x * 128 + (wid & 1) * 64;
  const int z = blockIdx.z;
  const int half = lane >> 4;                 // 0 / 1

  // Simple GEP chains off kernargs so addrspace inference keeps these global.
  const bf16_t* a0 = A + (long)z * strideA +
                     (long)(tileM + (lane & 15)) * lda + (half << 3);
  const bf16_t* a1 = a0 + 16 * lda;
  const bf16_t* b0 = Bt + (long)z * strideB +
                     (long)(tileN + (lane & 15)) * ldb + (half << 3);
  const bf16_t* b1 = b0 + 16 * ldb;
  const bf16_t* b2 = b0 + 32 * ldb;
  const bf16_t* b3 = b0 + 48 * ldb;

  const v8f zero = {0.f, 0.f, 0.f, 0.f, 0.f, 0.f, 0.f, 0.f};
  v8f acc[2][4];
#pragma unroll
  for (int mi = 0; mi < 2; ++mi)
#pragma unroll
    for (int ni = 0; ni < 4; ++ni) acc[mi][ni] = zero;

  for (int kb = 0; kb < K; kb += 32) {
    const v16bf af0 = ldA(a0 + kb);
    const v16bf af1 = ldA(a1 + kb);
    const v16bf bf0 = ldB(b0 + kb);
    const v16bf bf1 = ldB(b1 + kb);
    const v16bf bf2 = ldB(b2 + kb);
    const v16bf bf3 = ldB(b3 + kb);
    acc[0][0] = __builtin_amdgcn_wmma_f32_16x16x32_bf16(
        false, af0, false, bf0, (short)0, acc[0][0], false, false);
    acc[0][1] = __builtin_amdgcn_wmma_f32_16x16x32_bf16(
        false, af0, false, bf1, (short)0, acc[0][1], false, false);
    acc[0][2] = __builtin_amdgcn_wmma_f32_16x16x32_bf16(
        false, af0, false, bf2, (short)0, acc[0][2], false, false);
    acc[0][3] = __builtin_amdgcn_wmma_f32_16x16x32_bf16(
        false, af0, false, bf3, (short)0, acc[0][3], false, false);
    acc[1][0] = __builtin_amdgcn_wmma_f32_16x16x32_bf16(
        false, af1, false, bf0, (short)0, acc[1][0], false, false);
    acc[1][1] = __builtin_amdgcn_wmma_f32_16x16x32_bf16(
        false, af1, false, bf1, (short)0, acc[1][1], false, false);
    acc[1][2] = __builtin_amdgcn_wmma_f32_16x16x32_bf16(
        false, af1, false, bf2, (short)0, acc[1][2], false, false);
    acc[1][3] = __builtin_amdgcn_wmma_f32_16x16x32_bf16(
        false, af1, false, bf3, (short)0, acc[1][3], false, false);
  }

  // Epilogue.
#pragma unroll
  for (int mi = 0; mi < 2; ++mi) {
    const int rowBase = tileM + mi * 16 + (half << 3);
    // Batch split for OUT_MODE 2 (128-row tiles never straddle a batch).
    const int bo = (OUT_MODE == 2) ? (rowBase / Srows) : 0;
    const int s0 = (OUT_MODE == 2) ? (rowBase - bo * Srows) : 0;
#pragma unroll
    for (int ni = 0; ni < 4; ++ni) {
      const int col = tileN + ni * 16 + (lane & 15);
      const float bv = BIAS ? bias[col] : 0.0f;
#pragma unroll
      for (int r = 0; r < 8; ++r) {
        const float v = acc[mi][ni][r] + bv;
        if (OUT_MODE == 0) {
          float* C = (float*)Cp + (long)z * strideC;
          C[(long)(rowBase + r) * ldc + col] = v;
        } else if (OUT_MODE == 1) {
          bf16_t* C = (bf16_t*)Cp + (long)z * strideC;
          C[(long)(rowBase + r) * ldc + col] = (bf16_t)v;
        } else {
          bf16_t* C = (bf16_t*)Cp;            // [batch][N][Srows]
          C[(long)bo * N * (long)Srows + (long)col * Srows + (s0 + r)] = (bf16_t)v;
        }
      }
    }
  }
}

// ---------------------------------------------------------------------------
// Row softmax over S=4096: one 256-thread block per row, values in registers.
// Writes normalized f32 in place and a bf16 copy for the attn@V GEMM.
// ---------------------------------------------------------------------------
__global__ __launch_bounds__(256) void softmax_rows(float* __restrict__ attn,
                                                    bf16_t* __restrict__ attn_bf) {
  const long row = blockIdx.x;
  float*  p  = attn    + row * 4096l;
  bf16_t* pb = attn_bf + row * 4096l;
  const int tid = threadIdx.x, lane = tid & 31, wid = tid >> 5;
  __shared__ float red[8];

  float v[16];
#pragma unroll
  for (int j = 0; j < 16; ++j) v[j] = p[tid + j * 256];

  float m = v[0];
#pragma unroll
  for (int j = 1; j < 16; ++j) m = fmaxf(m, v[j]);
#pragma unroll
  for (int off = 16; off > 0; off >>= 1) m = fmaxf(m, __shfl_xor(m, off, 32));
  if (lane == 0) red[wid] = m;
  __syncthreads();
  float mall = red[0];
#pragma unroll
  for (int j = 1; j < 8; ++j) mall = fmaxf(mall, red[j]);
  __syncthreads();

  float s = 0.f;
#pragma unroll
  for (int j = 0; j < 16; ++j) { v[j] = __expf(v[j] - mall); s += v[j]; }
#pragma unroll
  for (int off = 16; off > 0; off >>= 1) s += __shfl_xor(s, off, 32);
  if (lane == 0) red[wid] = s;
  __syncthreads();
  float tot = 0.f;
#pragma unroll
  for (int j = 0; j < 8; ++j) tot += red[j];
  const float inv = 1.0f / tot;

#pragma unroll
  for (int j = 0; j < 16; ++j) {
    const float o = v[j] * inv;
    p[tid + j * 256]  = o;
    pb[tid + j * 256] = (bf16_t)o;
  }
}

// ---------------------------------------------------------------------------
extern "C" void kernel_launch(void* const* d_in, const int* in_sizes, int n_in,
                              void* d_out, int out_size, void* d_ws, size_t ws_size,
                              hipStream_t stream) {
  const float* query = (const float*)d_in[0];
  const float* key   = (const float*)d_in[1];
  const float* value = (const float*)d_in[2];
  const float* W     = (const float*)d_in[3];   // [D,D] = [out,in]
  const float* bias  = (const float*)d_in[4];   // [D]

  const int  Bn = 4, S = 4096, D = 1024;
  const long SD = (long)S * D;                  // 4,194,304
  const long SS = (long)S * S;                  // 16,777,216
  const long MS = (long)Bn * S;                 // 16,384 total rows
  const long DD = (long)D * D;                  // 1,048,576

  float* out_o = (float*)d_out;                 // [B,S,D]
  float* out_a = (float*)d_out + Bn * SD;       // [B,S,S] (attn, f32)

  // Workspace layout (~224 MB total). The converted-input slab is dead after
  // the projections, so the bf16 attn copy aliases it.
  bf16_t* a_bf  = (bf16_t*)d_ws;                // Bn*SS elems (134 MB), phase 2
  bf16_t* xq_bf = (bf16_t*)d_ws;                // phase 1 (aliases a_bf):
  bf16_t* xk_bf = xq_bf + Bn * SD;
  bf16_t* xv_bf = xk_bf + Bn * SD;
  bf16_t* W_bf  = xv_bf + Bn * SD;              //   ... + DD elems  (< Bn*SS)
  bf16_t* q_bf  = (bf16_t*)d_ws + Bn * SS;      // live to the end:
  bf16_t* k_bf  = q_bf + Bn * SD;
  bf16_t* vT_bf = k_bf + Bn * SD;               //   [B][D][S]

  const dim3 blk(256, 1, 1);

  // 0) One-shot f32 -> bf16 conversions (8 elems / thread).
  cvt_f32_bf16<<<dim3((unsigned)(Bn * SD / 2048)), blk, 0, stream>>>(query, xq_bf);
  cvt_f32_bf16<<<dim3((unsigned)(Bn * SD / 2048)), blk, 0, stream>>>(key,   xk_bf);
  cvt_f32_bf16<<<dim3((unsigned)(Bn * SD / 2048)), blk, 0, stream>>>(value, xv_bf);
  cvt_f32_bf16<<<dim3((unsigned)(DD / 2048)),      blk, 0, stream>>>(W,     W_bf);

  // 1) Projections: M = B*S rows, N = K = D, shared W/bias.
  const dim3 gProj(D / 128, (unsigned)(MS / 128), 1);
  wmma_gemm_bt<true, 1><<<gProj, blk, 0, stream>>>(
      xq_bf, W_bf, bias, q_bf, D, D, D, D, D, 0, 0, 0, S);
  wmma_gemm_bt<true, 1><<<gProj, blk, 0, stream>>>(
      xk_bf, W_bf, bias, k_bf, D, D, D, D, D, 0, 0, 0, S);
  wmma_gemm_bt<true, 2><<<gProj, blk, 0, stream>>>(
      xv_bf, W_bf, bias, vT_bf, D, D, D, D, D, 0, 0, 0, S);  // transposed bf16

  // 2) Scores: per batch S x S, K = D; raw f32 straight into d_out attn slab.
  const dim3 gScore(S / 128, S / 128, Bn);
  wmma_gemm_bt<false, 0><<<gScore, blk, 0, stream>>>(
      q_bf, k_bf, nullptr, out_a, S, D, D, D, S, SD, SD, SS, S);

  // 3) Softmax each of the B*S rows (in-place f32 + bf16 copy into a_bf).
  softmax_rows<<<dim3((unsigned)MS, 1, 1), blk, 0, stream>>>(out_a, a_bf);

  // 4) Output: per batch S x D, K = S; A = attn bf16, B = v^T bf16.
  const dim3 gOut(D / 128, S / 128, Bn);
  wmma_gemm_bt<false, 0><<<gOut, blk, 0, stream>>>(
      a_bf, vT_bf, nullptr, out_o, D, S, S, S, D, SS, SD, SD, S);
}
